// BubblePredictor_58205396795414
// MI455X (gfx1250) — matrix operand order repaired and checked by reading, not compiled, not run on hardware
//
#include <hip/hip_runtime.h>
#include <hip/hip_bf16.h>
#include <math.h>
#include <stdint.h>

typedef __attribute__((ext_vector_type(16))) __bf16 v16bf;
typedef __attribute__((ext_vector_type(8)))  float  v8f;

#define T_STEPS 1024
#define HID     512
#define NTILES  128   // 4H/16 = 2048/16
#define KCH     17    // 16 hidden K-chunks (512/32) + 1 input K-chunk (8 real + 24 zero)
#define HBUF    8192  // one h double-buffer: 16 chunks * 512 bf16
#define XEXT    16384 // x-extension tile base (elems)

#if defined(__has_builtin)
#if __has_builtin(__builtin_amdgcn_tanhf)
#define TANHF __builtin_amdgcn_tanhf
#endif
#endif
#ifndef TANHF
#define TANHF tanhf
#endif

__device__ __forceinline__ float sigmoid_fast(float x) {
  // sigmoid(x) = 0.5*tanh(x/2) + 0.5  -> single v_tanh_f32 on gfx1250
  return __builtin_fmaf(0.5f, TANHF(0.5f * x), 0.5f);
}

// ---------------------------------------------------------------------------
// Prep: pack fused weights [W_hh | W_ih | 0] (K = 512+32) into bf16 tiles in
// WMMA B-operand lane layout:  wt[ntile][kc][lane][16 elems], 1 KB per tile.
// B layout (16-bit, 32x16): lane L: n = L%16 ; elem e: k = (L/16)*8 + e + (e>=8 ? 8 : 0).
// ---------------------------------------------------------------------------
__global__ void lstm_pack_weights(const float* __restrict__ W_ih,
                                  const float* __restrict__ W_hh,
                                  __bf16* __restrict__ wt) {
  const int tile  = blockIdx.x;          // 0 .. NTILES*KCH-1
  const int ntile = tile / KCH;
  const int kc    = tile % KCH;
  const int lane  = threadIdx.x;         // 0..31
  const int n     = ntile * 16 + (lane & 15);
  const int khalf = (lane >> 4) * 8;
  __bf16* dst = wt + (size_t)tile * 512 + lane * 16;
#pragma unroll
  for (int e = 0; e < 16; ++e) {
    const int k = kc * 32 + khalf + e + (e >= 8 ? 8 : 0);
    float v = 0.f;
    if (k < HID) {
      v = W_hh[(size_t)n * HID + k];
    } else {
      const int i = k - HID;
      if (i < 8) v = W_ih[n * 8 + i];
    }
    dst[e] = (__bf16)v;
  }
}

// ---------------------------------------------------------------------------
// Persistent LSTM scan. One workgroup per 16-batch tile (grid = 16), 512
// threads = 16 waves. Wave w owns hidden slice j in [32w, 32w+32): its 8
// WMMA N-tiles are {gate g, j-subtile jt} -> ntile = g*32 + 2w + jt, so all
// four gates for its j's live in its own accumulators (register-local cell
// update). h is double-buffered in LDS in WMMA A-operand layout; the x
// projection is folded in as K-chunk 16 and prefetched one step ahead.
// Two barriers per step. Weights stream from L2 every step (2.18 MB bf16,
// permanently L2-resident).
// ---------------------------------------------------------------------------
__global__ void __launch_bounds__(512, 1)
lstm_scan(const float* __restrict__ history,
          const __bf16* __restrict__ wt,
          const float* __restrict__ W_head,
          const float* __restrict__ b_head,
          float* __restrict__ out) {   // [B*T*2 logits][B*H h][B*H c]
  __shared__ __attribute__((aligned(32))) __bf16 hA[2 * HBUF + 512];
  __shared__ float logit_s[32];

  const int tid   = threadIdx.x;
  const int w     = tid >> 5;        // wave 0..15
  const int lane  = tid & 31;
  const int n     = lane & 15;
  const int half  = lane >> 4;       // C-tile row half: m = half*8 + r
  const int btile = blockIdx.x;      // 0..15

  // zero-init both h buffers (t=0 state) and the x-ext pad region
  for (int i = tid; i < 2 * HBUF + 512; i += 512) hA[i] = (__bf16)0.f;
  if (w == 0) logit_s[lane] = 0.f;

  // Per-thread elementwise constants. This thread's (b,j) coverage:
  //   b = btile*16 + half*8 + r (r=0..7),  j = w*32 + jt*16 + n (jt=0,1)
  // A-operand scatter address for h[m = half*8+r][k_local = jt*16+n], chunk w:
  //   ha=(kl>>3)&1; hi=kl>>4; kk=kl&7; v=(kk>>1)+hi*4; e=2v+(kk&1);
  //   elem = w*512 + (ha*16 + m)*16 + e
  int   jj[2], c0[2];
  float wh0[2], wh1[2];
#pragma unroll
  for (int jt = 0; jt < 2; ++jt) {
    const int j  = w * 32 + jt * 16 + n;
    const int kl = jt * 16 + n;
    const int ha = (kl >> 3) & 1;
    const int hi = kl >> 4;
    const int kk = kl & 7;
    const int v  = (kk >> 1) + hi * 4;
    const int e  = 2 * v + (kk & 1);
    jj[jt]  = j;
    c0[jt]  = w * 512 + (ha * 16 + half * 8) * 16 + e;   // + r*16 per row
    wh0[jt] = W_head[j];
    wh1[jt] = W_head[HID + j];
  }
  const float bh0 = b_head[0], bh1 = b_head[1];

  float cst[16];
#pragma unroll
  for (int i = 0; i < 16; ++i) cst[i] = 0.f;

  // x-ext tile for t = 0 (chunk 16: lanes 0..15 rows, elems 0..7 = inputs)
  if (w == 15) {
    const int m  = lane & 15;
    const int ib = (lane >> 4) * 4;
    const float* xp = history + (((size_t)(btile * 16 + m)) * T_STEPS) * 8 + ib;
#pragma unroll
    for (int ii = 0; ii < 4; ++ii)
      hA[XEXT + m * 16 + ib + ii] = (__bf16)xp[ii];
  }
  __syncthreads();

  int p = 0;   // which h buffer GEMM reads this step
#pragma clang loop unroll(disable)
  for (int t = 0; t < T_STEPS; ++t) {
    // Opaque zero index, redefined every iteration: makes the 272 weight
    // load addresses formally loop-variant so LICM cannot hoist them out of
    // the t-loop (which caused scratch spills), while keeping the pointer's
    // global address-space provenance so they lower as global_load_b128
    // (LOADcnt only), not flat_load_b128 (LOADcnt+DScnt, aperture checks).
    uint32_t zoff = 0;
    asm volatile("" : "+s"(zoff));
    const v16bf* __restrict__ wtv = (const v16bf*)wt + zoff;

    // ---- gates = [h | x] @ [W_hh | W_ih]^T : 8 tiles x 17 K-chunks --------
    // Fully unrolled so the 8 f32 accumulator tiles keep fixed registers
    // (no loop-carried acc copies -> no WMMA->VALU hazard nops).
    v8f acc[8];
#pragma unroll
    for (int u = 0; u < 8; ++u) acc[u] = (v8f){0.f,0.f,0.f,0.f,0.f,0.f,0.f,0.f};

    const int pbase = p * HBUF;
#pragma unroll
    for (int kc = 0; kc < KCH; ++kc) {
      const int abase = (kc < 16) ? (pbase + kc * 512) : XEXT;
      const v16bf a = *(const v16bf*)(&hA[abase + lane * 16]);
#pragma unroll
      for (int u = 0; u < 8; ++u) {
        const int ntile = (u >> 1) * 32 + w * 2 + (u & 1);  // gate g, subtile jt
        const v16bf b = wtv[(size_t)(ntile * KCH + kc) * 32 + lane];
        acc[u] = __builtin_amdgcn_wmma_f32_16x16x32_bf16(
            false, a, false, b, (short)0, acc[u], false, false);
      }
    }
    __syncthreads();   // all A-reads of buffer p / xext(t) done

    // ---- elementwise LSTM cell update (all 4 gates register-local) --------
    const int wbase = (p ^ 1) * HBUF;
#pragma unroll
    for (int r = 0; r < 8; ++r) {
      float p0 = 0.f, p1 = 0.f;
#pragma unroll
      for (int jt = 0; jt < 2; ++jt) {
        const float iv = acc[0 + jt][r];
        const float fv = acc[2 + jt][r];
        const float gv = acc[4 + jt][r];
        const float ov = acc[6 + jt][r];
        const int   ci = jt * 8 + r;
        const float c  = sigmoid_fast(fv) * cst[ci] +
                         sigmoid_fast(iv) * TANHF(gv);
        cst[ci] = c;
        const float h = sigmoid_fast(ov) * TANHF(c);
        hA[wbase + c0[jt] + r * 16] = (__bf16)h;   // scatter into A layout
        p0 += h * wh0[jt];
        p1 += h * wh1[jt];
        if (t == T_STEPS - 1) {                    // final (h, c), fp32
          const size_t bg = (size_t)(btile * 16 + half * 8 + r);
          out[524288 + bg * HID + jj[jt]] = h;
          out[524288 + 131072 + bg * HID + jj[jt]] = c;
        }
      }
      const int bo = (half * 8 + r) * 2;
      atomicAdd(&logit_s[bo],     p0);             // ds_add_f32
      atomicAdd(&logit_s[bo + 1], p1);
    }

    // prefetch x(t+1) into the x-ext tile (no one reads it until next GEMM)
    if (w == 15 && t + 1 < T_STEPS) {
      const int m  = lane & 15;
      const int ib = (lane >> 4) * 4;
      const float* xp =
          history + (((size_t)(btile * 16 + m)) * T_STEPS + (t + 1)) * 8 + ib;
#pragma unroll
      for (int ii = 0; ii < 4; ++ii)
        hA[XEXT + m * 16 + ib + ii] = (__bf16)xp[ii];
    }
    __syncthreads();   // h(t), x(t+1), logit atomics all visible

    if (w == 0) {
      // drain this step's logits; re-zero before next step's atomics
      const int b = btile * 16 + (lane >> 1);
      const int o = lane & 1;
      out[((size_t)b * T_STEPS + t) * 2 + o] = logit_s[lane] + (o ? bh1 : bh0);
      logit_s[lane] = 0.f;
    }
    p ^= 1;
  }
}

// ---------------------------------------------------------------------------
extern "C" void kernel_launch(void* const* d_in, const int* in_sizes, int n_in,
                              void* d_out, int out_size, void* d_ws,
                              size_t ws_size, hipStream_t stream) {
  const float* history = (const float*)d_in[0];   // [256,1024,8]
  const float* W_ih    = (const float*)d_in[1];   // [2048,8]
  const float* W_hh    = (const float*)d_in[2];   // [2048,512]
  const float* W_head  = (const float*)d_in[3];   // [2,512]
  const float* b_head  = (const float*)d_in[4];   // [2]
  float* out = (float*)d_out;                     // logits | h | c
  __bf16* wt = (__bf16*)d_ws;                     // 128*17*512 bf16 = 2.18 MB

  lstm_pack_weights<<<NTILES * KCH, 32, 0, stream>>>(W_ih, W_hh, wt);
  lstm_scan<<<16, 512, 0, stream>>>(history, wt, W_head, b_head, out);
}
